// VectorQuantizer_5437428597119
// MI455X (gfx1250) — matrix operand order repaired and checked by reading, not compiled, not run on hardware
//
#include <hip/hip_runtime.h>
#include <hip/hip_bf16.h>

// ---------------------------------------------------------------------------
// VQ eval: argmin_k ||x_n - w_k||^2, gather winners, mismatch loss.
// N=8192 latents, K=8192 codes, D=512.
// score(n,k) = ||w_k||^2 - 2 * x_n . w_k   (||x_n||^2 dropped: row-constant)
// GEMM X.W^T with bf16 hi/lo split (3x v_wmma_f32_16x16x32_bf16 per tile/step)
// W operand staged in LDS per workgroup via async global->LDS copies
// (double-buffered, ASYNCcnt-tracked) and shared by all 8 waves.
// ---------------------------------------------------------------------------

#define NROW 8192   // N latents
#define KCB  8192   // K codebook entries
#define DD   512    // embedding dim

typedef __attribute__((ext_vector_type(16))) __bf16       v16bf;
typedef __attribute__((ext_vector_type(4)))  __bf16       v4bf;
typedef __attribute__((ext_vector_type(8)))  float        v8f;
typedef __attribute__((ext_vector_type(4)))  unsigned int u32x4;
typedef __attribute__((ext_vector_type(8)))  unsigned int u32x8;
typedef __attribute__((ext_vector_type(4)))  int          i32x4;

#define AS_GLOBAL __attribute__((address_space(1)))
#define AS_LOCAL  __attribute__((address_space(3)))

#if defined(__has_builtin)
#if __has_builtin(__builtin_amdgcn_global_load_async_to_lds_b128)
#define HAVE_ASYNC_LDS 1
#endif
#endif
#ifndef HAVE_ASYNC_LDS
#define HAVE_ASYNC_LDS 0
#endif

__device__ inline void wait_async_le2() {
#if __has_builtin(__builtin_amdgcn_s_wait_asynccnt)
    __builtin_amdgcn_s_wait_asynccnt(2);
#else
    asm volatile("s_wait_asynccnt 0x2" ::: "memory");
#endif
}
__device__ inline void wait_async_le0() {
#if __has_builtin(__builtin_amdgcn_s_wait_asynccnt)
    __builtin_amdgcn_s_wait_asynccnt(0);
#else
    asm volatile("s_wait_asynccnt 0x0" ::: "memory");
#endif
}

// Load a 16-element bf16 WMMA fragment from two 16-byte chunks (global or LDS).
__device__ inline v16bf load_frag2(const __bf16* p0, const __bf16* p1) {
    u32x4 a = *(const u32x4*)p0;
    u32x4 b = *(const u32x4*)p1;
    u32x8 t;
    t[0] = a[0]; t[1] = a[1]; t[2] = a[2]; t[3] = a[3];
    t[4] = b[0]; t[5] = b[1]; t[6] = b[2]; t[7] = b[3];
    return __builtin_bit_cast(v16bf, t);
}

// ---------------------------------------------------------------------------
// Kernel 1: split X (blocks [0,8192)) and W (blocks [8192,16384)) into bf16
// hi/lo planes; for W rows also compute ||w_k||^2. Resets the loss flag.
// ---------------------------------------------------------------------------
__global__ void __launch_bounds__(128)
vq_prep(const float* __restrict__ X, const float* __restrict__ W,
        __bf16* __restrict__ XH, __bf16* __restrict__ XL,
        __bf16* __restrict__ WH, __bf16* __restrict__ WL,
        float* __restrict__ wn, int* __restrict__ flag)
{
    const int blk = blockIdx.x;
    const int t   = threadIdx.x;
    if (blk == 0 && t == 0) *flag = 0;

    const bool isW = blk >= NROW;
    const int  row = isW ? (blk - NROW) : blk;
    const float* src = (isW ? W : X) + (size_t)row * DD;
    __bf16* dh = (isW ? WH : XH) + (size_t)row * DD;
    __bf16* dl = (isW ? WL : XL) + (size_t)row * DD;

    float4 v = *(const float4*)(src + t * 4);
    float e[4] = {v.x, v.y, v.z, v.w};
    v4bf h, l;
    float ss = 0.0f;
#pragma unroll
    for (int i = 0; i < 4; ++i) {
        __bf16 hi = (__bf16)e[i];
        h[i] = hi;
        l[i] = (__bf16)(e[i] - (float)hi);
        ss += e[i] * e[i];
    }
    *(v4bf*)(dh + t * 4) = h;
    *(v4bf*)(dl + t * 4) = l;

    if (isW) {
#pragma unroll
        for (int o = 16; o >= 1; o >>= 1) ss += __shfl_down(ss, o, 32);
        __shared__ float red[4];
        if ((t & 31) == 0) red[t >> 5] = ss;
        __syncthreads();
        if (t == 0) wn[row] = red[0] + red[1] + red[2] + red[3];
    }
}

// ---------------------------------------------------------------------------
// Kernel 2: fused GEMM + running argmin.
// grid = (64 row-blocks, 8 K-chunks), 256 threads (8 waves).
// Per chunk: 16 column blocks of 64 x 16 k-steps of 32 = 256 slabs.
// Each slab of W (64 cols x 32 d, hi+lo planes, 8 KB) is async-copied to a
// double-buffered LDS stage once per workgroup and consumed by all 8 waves
// as B fragments (ds_load_b128). A fragments are row-private global loads.
// ---------------------------------------------------------------------------
__global__ void __launch_bounds__(256)
vq_dist_argmin(const __bf16* __restrict__ XH, const __bf16* __restrict__ XL,
               const __bf16* __restrict__ WH, const __bf16* __restrict__ WL,
               const float* __restrict__ wn,
               float* __restrict__ partV, int* __restrict__ partI)
{
    __shared__ __bf16 sW[2][2][64][32];   // [buf][plane][col][d] = 16 KB

    const int rb   = blockIdx.x;          // 0..63  (row block of 128)
    const int ch   = blockIdx.y;          // 0..7   (1024-column chunk)
    const int tid  = threadIdx.x;
    const int wv   = tid >> 5;            // 0..7
    const int lane = tid & 31;
    const int lrow = lane & 15;
    const int lhi  = lane >> 4;

    const int rstart = rb * 128 + wv * 16;
    const __bf16* xh = XH + (size_t)(rstart + lrow) * DD;
    const __bf16* xl = XL + (size_t)(rstart + lrow) * DD;
    const int cbase = ch * 1024;

    // Stage one 8 KB W slab (hi+lo) into LDS: 512 x 16B chunks, 2 per thread.
    auto fill = [&](int slab, int buf) {
        const int d0   = (slab & 15) * 32;
        const int colb = cbase + (slab >> 4) * 64;
#pragma unroll
        for (int u = 0; u < 2; ++u) {
            const int c     = tid + u * 256;
            const int plane = c >> 8;          // 0 = hi, 1 = lo
            const int col   = (c >> 2) & 63;
            const int seg   = c & 3;
            const __bf16* src = (plane ? WL : WH)
                                + (size_t)(colb + col) * DD + d0 + seg * 8;
            __bf16* dst = &sW[buf][plane][col][seg * 8];
#if HAVE_ASYNC_LDS
            __builtin_amdgcn_global_load_async_to_lds_b128(
                (AS_GLOBAL i32x4*)const_cast<__bf16*>(src),
                (AS_LOCAL  i32x4*)dst, 0, 0);
#else
            *(u32x4*)dst = *(const u32x4*)src;
#endif
        }
    };

    float bestv[8];
    int   besti[8];
#pragma unroll
    for (int i = 0; i < 8; ++i) { bestv[i] = 3.4e38f; besti[i] = 0; }

    v8f acc[4];
    const int SLABS = 256;                 // 16 col-blocks x 16 k-steps
    fill(0, 0);

    for (int s = 0; s < SLABS; ++s) {
        const int buf  = s & 1;
        const int d0   = (s & 15) * 32;
        const int it   = s >> 4;
        const int colb = cbase + it * 64;

        // Kick the next slab into the other buffer, then wait for this one.
        if (s + 1 < SLABS) {
            fill(s + 1, buf ^ 1);
#if HAVE_ASYNC_LDS
            wait_async_le2();
#endif
        } else {
#if HAVE_ASYNC_LDS
            wait_async_le0();
#endif
        }
        __syncthreads();                   // slab s visible to all waves

        if ((s & 15) == 0) {
#pragma unroll
            for (int t = 0; t < 4; ++t) { v8f z = {}; acc[t] = z; }
            // Uniform speculative prefetch of the next column block of W-hi.
            if (it < 15)
                __builtin_prefetch(WH + (size_t)(colb + 64 + lrow) * DD, 0, 0);
        }

        // A fragment (16x32 bf16): lanes 0-15 hold K 0..15,
        // lanes 16-31 hold K 16..31 of this 32-wide slab.
        v16bf ah = load_frag2(xh + d0 + lhi * 16, xh + d0 + lhi * 16 + 8);
        v16bf al = load_frag2(xl + d0 + lhi * 16, xl + d0 + lhi * 16 + 8);
#pragma unroll
        for (int t = 0; t < 4; ++t) {
            // B fragment (32x16 bf16): lane = column, halves split K 0..15 /
            // 16..31 -> 32 contiguous bf16 of the staged W row in LDS.
            const __bf16* bh = &sW[buf][0][t * 16 + lrow][lhi * 16];
            const __bf16* bl = &sW[buf][1][t * 16 + lrow][lhi * 16];
            v16bf fbh = load_frag2(bh, bh + 8);
            v16bf fbl = load_frag2(bl, bl + 8);
            acc[t] = __builtin_amdgcn_wmma_f32_16x16x32_bf16(
                false, ah, false, fbh, (short)0, acc[t], false, false);
            acc[t] = __builtin_amdgcn_wmma_f32_16x16x32_bf16(
                false, ah, false, fbl, (short)0, acc[t], false, false);
            acc[t] = __builtin_amdgcn_wmma_f32_16x16x32_bf16(
                false, al, false, fbh, (short)0, acc[t], false, false);
        }

        // End of a column block: fold tiles into the running argmin.
        // C layout: slot i holds row (i + 8*lhi), column (colb + t*16 + lrow).
        if ((s & 15) == 15) {
#pragma unroll
            for (int t = 0; t < 4; ++t) {
                const int col = colb + t * 16 + lrow;
                const float wnc = wn[col];
#pragma unroll
                for (int i = 0; i < 8; ++i) {
                    float sc = wnc - 2.0f * acc[t][i];
                    if (sc < bestv[i]) { bestv[i] = sc; besti[i] = col; }
                }
            }
        }
        __syncthreads();                   // protect buf before it is refilled
    }

    // Reduce across the 16 lanes that share each row (xor stays in-half).
#pragma unroll
    for (int i = 0; i < 8; ++i) {
        float bv = bestv[i];
        int   bi = besti[i];
#pragma unroll
        for (int o = 8; o >= 1; o >>= 1) {
            float ov = __shfl_xor(bv, o, 32);
            int   oi = __shfl_xor(bi, o, 32);
            if (ov < bv || (ov == bv && oi < bi)) { bv = ov; bi = oi; }
        }
        if (lrow == 0) {
            const int row = rstart + i + 8 * lhi;
            partV[(size_t)ch * NROW + row] = bv;
            partI[(size_t)ch * NROW + row] = bi;
        }
    }
}

// ---------------------------------------------------------------------------
// Kernel 3: reduce the 8 chunk partials per row -> final index; mismatch flag.
// ---------------------------------------------------------------------------
__global__ void __launch_bounds__(256)
vq_finalize(const float* __restrict__ partV, const int* __restrict__ partI,
            const int* __restrict__ label, int* __restrict__ idxOut,
            int* __restrict__ flag)
{
    const int n = blockIdx.x * blockDim.x + threadIdx.x;
    if (n >= NROW) return;
    float bv = 3.4e38f;
    int   bi = 0;
#pragma unroll
    for (int c = 0; c < 8; ++c) {
        float v  = partV[(size_t)c * NROW + n];
        int   ix = partI[(size_t)c * NROW + n];
        if (v < bv) { bv = v; bi = ix; }   // chunks scanned low->high cols
    }
    idxOut[n] = bi;
    if (bi != label[n]) atomicOr(flag, 1);
}

// ---------------------------------------------------------------------------
// Kernel 4: gather winning codebook rows (fp32) + scalar loss.
// ---------------------------------------------------------------------------
__global__ void __launch_bounds__(128)
vq_gather(const float* __restrict__ W, const int* __restrict__ idxOut,
          const int* __restrict__ flag, float* __restrict__ out)
{
    const int n = blockIdx.x;
    const int t = threadIdx.x;
    const int k = idxOut[n];
    const float4* src = (const float4*)(W + (size_t)k * DD);
    float4*       dst = (float4*)(out + (size_t)n * DD);
    dst[t] = src[t];
    if (n == 0 && t == 0)
        out[(size_t)NROW * DD] = (*flag) ? (float)NROW : 0.0f;
}

// ---------------------------------------------------------------------------
extern "C" void kernel_launch(void* const* d_in, const int* in_sizes, int n_in,
                              void* d_out, int out_size, void* d_ws, size_t ws_size,
                              hipStream_t stream)
{
    const float* X     = (const float*)d_in[0];  // flat_latents [N, D] f32
    const int*   label = (const int*)d_in[1];    // [N] i32
    const float* W     = (const float*)d_in[2];  // embedding_weight [K, D] f32

    char* ws = (char*)d_ws;
    const size_t planeB = (size_t)NROW * DD * sizeof(__bf16);  // 8 MB
    __bf16* XH   = (__bf16*)(ws);
    __bf16* XL   = (__bf16*)(ws + planeB);
    __bf16* WH   = (__bf16*)(ws + 2 * planeB);
    __bf16* WL   = (__bf16*)(ws + 3 * planeB);
    float*  wn   = (float*)(ws + 4 * planeB);
    float*  pV   = (float*)(ws + 4 * planeB + (size_t)KCB * 4);
    int*    pI   = (int*)  (ws + 4 * planeB + (size_t)KCB * 4 + (size_t)8 * NROW * 4);
    int*    idxO = (int*)  (ws + 4 * planeB + (size_t)KCB * 4 + (size_t)16 * NROW * 4);
    int*    flag = (int*)  (ws + 4 * planeB + (size_t)KCB * 4 + (size_t)17 * NROW * 4);

    vq_prep<<<2 * NROW, 128, 0, stream>>>(X, W, XH, XL, WH, WL, wn, flag);

    dim3 grid(NROW / 128, KCB / 1024);  // (64, 8)
    vq_dist_argmin<<<grid, 256, 0, stream>>>(XH, XL, WH, WL, wn, pV, pI);

    vq_finalize<<<NROW / 256, 256, 0, stream>>>(pV, pI, label, idxO, flag);

    vq_gather<<<NROW, 128, 0, stream>>>(W, idxO, flag, (float*)d_out);
}
